// Seq2Seq_82592221102723
// MI455X (gfx1250) — matrix operand order repaired and checked by reading, not compiled, not run on hardware
//
#include <hip/hip_runtime.h>

#define HDIM   128
#define LNUM   3
#define TSTEPS 500
#define GDIM   512      // 4*H
#define MT     16       // batch rows per workgroup

typedef __attribute__((ext_vector_type(16))) __bf16        bf16x16;
typedef __attribute__((ext_vector_type(8)))  float         f32x8;
typedef __attribute__((ext_vector_type(8)))  unsigned short u16x8;

// global-address-space pointers so weight loads lower to GLOBAL_LOAD (LOADcnt
// only), not FLAT (which also bumps DScnt and falsely orders vs LDS traffic)
typedef const unsigned short __attribute__((address_space(1)))* gwptr;
typedef const u16x8        __attribute__((address_space(1)))* gvptr;

union Frag { bf16x16 bf; u16x8 u[2]; };

__device__ __forceinline__ unsigned short f2bf(float f) {
  unsigned u = __float_as_uint(f);
  u += 0x7FFFu + ((u >> 16) & 1u);            // round-to-nearest-even
  return (unsigned short)(u >> 16);
}
__device__ __forceinline__ float bf2f(unsigned short b) {
  return __uint_as_float(((unsigned)b) << 16);
}
__device__ __forceinline__ float sigm(float x) { return 1.0f / (1.0f + __expf(-x)); }
__device__ __forceinline__ float tanh_f(float x) {
  x = fminf(12.0f, fmaxf(-12.0f, x));
  float e = __expf(2.0f * x);
  return (e - 1.0f) / (e + 1.0f);
}

// Defeat LICM: make a uniform pointer opaque so loop-invariant weight loads
// cannot be hoisted out of the 500-step time loop (which caused scratch
// spills).  Returns an address_space(1) pointer -> GLOBAL_LOAD codegen.
__device__ __forceinline__ gwptr launder_g(const unsigned short* p) {
  unsigned long long v = (unsigned long long)p;
  asm volatile("" : "+s"(v));                 // keep in SGPR pair, value unchanged
  return (gwptr)v;
}

// ---------------------------------------------------------------------------
// Repack fp32 weight matrices [512,128] into CDNA5 WMMA bf16 B-fragment layout.
// Fragment (nt,kt) covers N=nt*16..+16, K=kt*32..+32.  Within a fragment:
//   lane l holds N = nt*16 + (l&15);  half hi -> K = kt*32 + (l>=16)*8 + (hi&7) + (hi>=8)*16.
// Stored as [matrix][frag = nt*4+kt][lane][hi] -> 512 halves (1KB) per fragment.
// 10 matrices: encWih1,encWih2,encWhh0..2,decWih1,decWih2,decWhh0..2
// ---------------------------------------------------------------------------
__global__ __launch_bounds__(256) void repack_weights(
    const float* __restrict__ enc_Wih12, const float* __restrict__ enc_Whh,
    const float* __restrict__ dec_Wih12, const float* __restrict__ dec_Whh,
    unsigned short* __restrict__ out)
{
  int gid  = blockIdx.x * 256 + threadIdx.x;   // 10 * 65536 total
  int mat  = gid >> 16;
  int rem  = gid & 0xFFFF;
  int frag = rem >> 9;
  int lane = (rem >> 4) & 31;
  int hi   = rem & 15;
  int nt = frag >> 2, kt = frag & 3;
  int n = nt * 16 + (lane & 15);
  int k = kt * 32 + ((lane >> 4) << 3) + (hi & 7) + ((hi >> 3) << 4);
  const float* srcm;
  switch (mat) {
    case 0: srcm = enc_Wih12;           break;
    case 1: srcm = enc_Wih12 + 65536;   break;
    case 2: srcm = enc_Whh;             break;
    case 3: srcm = enc_Whh + 65536;     break;
    case 4: srcm = enc_Whh + 131072;    break;
    case 5: srcm = dec_Wih12;           break;
    case 6: srcm = dec_Wih12 + 65536;   break;
    case 7: srcm = dec_Whh;             break;
    case 8: srcm = dec_Whh + 65536;     break;
    default: srcm = dec_Whh + 131072;   break;
  }
  out[gid] = f2bf(srcm[n * HDIM + k]);
}

// A fragment (16x32 bf16) from LDS h rows (padded stride HDIM+8 halves).
// lanes 0-15: M=lane, halves 0-7 -> K=k0..k0+7, halves 8-15 -> K=k0+16..k0+23
// lanes 16-31: same M, K offset +8 / +24.
__device__ __forceinline__ bf16x16 load_a(const unsigned short* base, int lane, int k0) {
  const unsigned short* p = base + (lane & 15) * (HDIM + 8) + k0 + ((lane >> 4) << 3);
  Frag a;
  a.u[0] = *(const u16x8*)(p);
  a.u[1] = *(const u16x8*)(p + 16);
  return a.bf;
}

// B fragment (32x16 bf16) from repacked global weights: 2x global_load_b128.
__device__ __forceinline__ bf16x16 load_b(gwptr wmat, int nt, int kt, int lane) {
  gwptr p = wmat + (((nt << 2) + kt) << 9) + (lane << 4);
  Frag b;
  b.u[0] = *(gvptr)(p);
  b.u[1] = *(gvptr)(p + 8);
  return b.bf;
}

// acc[t] += A(h, K) x B(w, nt=wnt+t, K) over K=128  (16 WMMAs)
__device__ __forceinline__ void mma4(f32x8 acc[4], const unsigned short* aBase,
                                     gwptr wmat, int wnt, int lane) {
#pragma unroll
  for (int kt = 0; kt < 4; ++kt) {
    bf16x16 a = load_a(aBase, lane, kt * 32);
#pragma unroll
    for (int t = 0; t < 4; ++t) {
      bf16x16 b = load_b(wmat, wnt + t, kt, lane);
      acc[t] = __builtin_amdgcn_wmma_f32_16x16x32_bf16(
          false, a, false, b, (short)0, acc[t], false, false);
    }
  }
}

// ---------------------------------------------------------------------------
// Persistent seq2seq kernel: 32 blocks x 16 batch rows, 8 waves x 64 gate cols.
// ---------------------------------------------------------------------------
__global__ __launch_bounds__(256) void seq2seq_lstm(
    const float* __restrict__ src,
    const float* __restrict__ enc_Wih0,
    const float* __restrict__ enc_bih, const float* __restrict__ enc_bhh,
    const float* __restrict__ dec_Wih0,
    const float* __restrict__ dec_bih, const float* __restrict__ dec_bhh,
    const float* __restrict__ fc_W,    const float* __restrict__ fc_b,
    const unsigned short* __restrict__ wpack,
    float* __restrict__ out)
{
  __shared__ float          gates[MT][GDIM + 4];           // padded stride breaks bank conflicts
  __shared__ unsigned short hsm[LNUM][MT][HDIM + 8];       // bf16 hidden state
  __shared__ float          csm[LNUM][MT][HDIM + 1];       // f32 cell state
  __shared__ float          xbuf[MT][2];
  __shared__ float          fcw[2 * HDIM];
  __shared__ float          fcb_s[2];

  const int tid  = threadIdx.x;
  const int lane = tid & 31;
  const int wave = tid >> 5;
  const int b0   = blockIdx.x * MT;

  for (int i = tid; i < LNUM * MT * (HDIM + 8); i += 256) ((unsigned short*)hsm)[i] = 0;
  for (int i = tid; i < LNUM * MT * (HDIM + 1); i += 256) ((float*)csm)[i] = 0.0f;
  if (tid < 2 * HDIM) fcw[tid] = fc_W[tid];
  if (tid < 2)        fcb_s[tid] = fc_b[tid];

  // per-wave, per-N-tile constants kept in registers for the whole run
  float biasE[LNUM][4], biasD[LNUM][4];
  float w0E[4][2], w0D[4][2];
#pragma unroll
  for (int t = 0; t < 4; ++t) {
    int n = wave * 64 + t * 16 + (lane & 15);
#pragma unroll
    for (int l = 0; l < LNUM; ++l) {
      biasE[l][t] = enc_bih[l * GDIM + n] + enc_bhh[l * GDIM + n];
      biasD[l][t] = dec_bih[l * GDIM + n] + dec_bhh[l * GDIM + n];
    }
    w0E[t][0] = enc_Wih0[n * 2 + 0]; w0E[t][1] = enc_Wih0[n * 2 + 1];
    w0D[t][0] = dec_Wih0[n * 2 + 0]; w0D[t][1] = dec_Wih0[n * 2 + 1];
  }

  const unsigned short* eWih[LNUM] = { nullptr,          wpack + 0 * 65536, wpack + 1 * 65536 };
  const unsigned short* eWhh[LNUM] = { wpack + 2 * 65536, wpack + 3 * 65536, wpack + 4 * 65536 };
  const unsigned short* dWih[LNUM] = { nullptr,          wpack + 5 * 65536, wpack + 6 * 65536 };
  const unsigned short* dWhh[LNUM] = { wpack + 7 * 65536, wpack + 8 * 65536, wpack + 9 * 65536 };

  __syncthreads();

  auto step = [&](const unsigned short* const* wih, const unsigned short* const* whh,
                  const float (*biasv)[4], const float (*w0)[2]) {
#pragma unroll
    for (int l = 0; l < LNUM; ++l) {
      f32x8 acc[4];
#pragma unroll
      for (int t = 0; t < 4; ++t)
#pragma unroll
        for (int r = 0; r < 8; ++r) acc[t][r] = biasv[l][t];

      if (l == 0) {
        // K=2 input matmul: VALU (WMMA would waste 94% of K)
        float x0[8], x1[8];
#pragma unroll
        for (int r = 0; r < 8; ++r) {
          int m = r + ((lane >> 4) << 3);
          x0[r] = xbuf[m][0]; x1[r] = xbuf[m][1];
        }
#pragma unroll
        for (int t = 0; t < 4; ++t)
#pragma unroll
          for (int r = 0; r < 8; ++r)
            acc[t][r] += x0[r] * w0[t][0] + x1[r] * w0[t][1];
      } else {
        // laundered global pointer: loads stay inside the time loop
        // (L2-resident streaming), never hoisted + spilled to scratch
        mma4(acc, &hsm[l - 1][0][0], launder_g(wih[l]), wave * 4, lane);
      }
      mma4(acc, &hsm[l][0][0], launder_g(whh[l]), wave * 4, lane);

      // C layout: vgpr r, lane -> M = r + 8*(lane>=16), N = lane&15
#pragma unroll
      for (int t = 0; t < 4; ++t)
#pragma unroll
        for (int r = 0; r < 8; ++r)
          gates[r + ((lane >> 4) << 3)][wave * 64 + t * 16 + (lane & 15)] = acc[t][r];
      __syncthreads();

      // elementwise cell update: 2048 cells / 256 threads
      {
        int m = tid & 15, jb = (tid >> 4) << 3;
#pragma unroll
        for (int jj = 0; jj < 8; ++jj) {
          int j = jb + jj;
          float ig = sigm(gates[m][j]);
          float fg = sigm(gates[m][HDIM + j]);
          float gg = tanh_f(gates[m][2 * HDIM + j]);
          float og = sigm(gates[m][3 * HDIM + j]);
          float cn = fg * csm[l][m][j] + ig * gg;
          float hn = og * tanh_f(cn);
          csm[l][m][j] = cn;
          hsm[l][m][j] = f2bf(hn);
        }
      }
      __syncthreads();
    }
  };

  // ---------------- encoder ----------------
#pragma unroll 1
  for (int t = 0; t < TSTEPS; ++t) {
    if (tid < 32) {
      int m = tid >> 1, c = tid & 1;
      xbuf[m][c] = src[((size_t)(b0 + m) * TSTEPS + t) * 2 + c];
    }
    __syncthreads();
    step(eWih, eWhh, biasE, w0E);
  }

  // x0 = src[:, -1, :]
  if (tid < 32) {
    int m = tid >> 1, c = tid & 1;
    xbuf[m][c] = src[((size_t)(b0 + m) * TSTEPS + (TSTEPS - 1)) * 2 + c];
  }
  __syncthreads();

  // ---------------- decoder ----------------
#pragma unroll 1
  for (int t = 0; t < TSTEPS; ++t) {
    step(dWih, dWhh, biasD, w0D);

    // pred = h_top @ fc_W^T + fc_b  (wave-parallel partial dot + LDS reduce)
    {
      int m = tid & 15, od = (tid >> 4) & 1, seg = tid >> 5;
      float p = 0.0f;
      int kb = seg * 16;
#pragma unroll
      for (int kk = 0; kk < 16; ++kk)
        p += bf2f(hsm[2][m][kb + kk]) * fcw[od * HDIM + kb + kk];
      float* part = &gates[0][0];               // reuse gates LDS as scratch
      part[(((od << 4) | m) << 3) | seg] = p;
      __syncthreads();
      if (tid < 32) {
        int m2 = tid & 15, od2 = tid >> 4;
        float s = fcb_s[od2];
        const float* pp = &gates[0][0] + (((od2 << 4) | m2) << 3);
#pragma unroll
        for (int q = 0; q < 8; ++q) s += pp[q];
        out[((size_t)(b0 + m2) * TSTEPS + t) * 2 + od2] = s;
        xbuf[m2][od2] = s;                      // autoregressive feedback
      }
      __syncthreads();
    }
  }
}

extern "C" void kernel_launch(void* const* d_in, const int* in_sizes, int n_in,
                              void* d_out, int out_size, void* d_ws, size_t ws_size,
                              hipStream_t stream) {
  (void)in_sizes; (void)n_in; (void)out_size; (void)ws_size;
  const float* src       = (const float*)d_in[0];
  const float* enc_Wih0  = (const float*)d_in[1];
  const float* enc_Wih12 = (const float*)d_in[2];
  const float* enc_Whh   = (const float*)d_in[3];
  const float* enc_bih   = (const float*)d_in[4];
  const float* enc_bhh   = (const float*)d_in[5];
  const float* dec_Wih0  = (const float*)d_in[6];
  const float* dec_Wih12 = (const float*)d_in[7];
  const float* dec_Whh   = (const float*)d_in[8];
  const float* dec_bih   = (const float*)d_in[9];
  const float* dec_bhh   = (const float*)d_in[10];
  const float* fc_W      = (const float*)d_in[11];
  const float* fc_b      = (const float*)d_in[12];
  unsigned short* wpack  = (unsigned short*)d_ws;   // 10*65536*2B = 1.25 MB

  repack_weights<<<2560, 256, 0, stream>>>(enc_Wih12, enc_Whh, dec_Wih12, dec_Whh, wpack);
  seq2seq_lstm<<<32, 256, 0, stream>>>(src, enc_Wih0, enc_bih, enc_bhh,
                                       dec_Wih0, dec_bih, dec_bhh,
                                       fc_W, fc_b, wpack, (float*)d_out);
}
